// MultiHeadSelfAttention_40690520162747
// MI455X (gfx1250) — matrix operand order repaired and checked by reading, not compiled, not run on hardware
//
#include <hip/hip_runtime.h>

#define BATCH    2
#define S_LEN    2048
#define D_MODEL  1024
#define N_HEADS  16
#define HEAD_DIM 64

typedef __attribute__((ext_vector_type(16))) _Float16 v16h;
typedef __attribute__((ext_vector_type(8)))  float    v8f;

union FragU { uint4 u[2]; v16h v; };

// A-matrix 16x32 f16 fragment (ISA 7.12.2): lane holds row M=lane&15;
// g=lane>>4: VGPR0-3 hold K = g*8 .. g*8+7, VGPR4-7 hold K = 16+g*8 .. +7.
__device__ __forceinline__ v16h load_frag_a(const _Float16* base, int g) {
  FragU f;
  f.u[0] = *(const uint4*)(base + g * 8);
  f.u[1] = *(const uint4*)(base + 16 + g * 8);
  return f.v;
}

// B-matrix 32x16 f16 fragment: lane holds column N=lane&15;
// half-wave g: VGPR0-3 hold K = g*8 .. g*8+7, VGPR4-7 hold K = 16+g*8 .. +7.
__device__ __forceinline__ v16h load_frag_b(const _Float16* base, int g) {
  FragU f;
  f.u[0] = *(const uint4*)(base + g * 8);
  f.u[1] = *(const uint4*)(base + 16 + g * 8);
  return f.v;
}

// CDNA5 async copy: 16B global -> LDS, tracked by ASYNCcnt.
__device__ __forceinline__ void async_ld16(const _Float16* gsrc, _Float16* ldst) {
  unsigned loff = (unsigned)(size_t)ldst;   // low 32 bits of shared ptr = LDS byte addr
  asm volatile("global_load_async_to_lds_b128 %0, %1, off"
               :: "v"(loff), "v"(gsrc) : "memory");
}
__device__ __forceinline__ void wait_async0() {
  asm volatile("s_wait_asynccnt 0x0" ::: "memory");
}

__global__ void f32_to_f16_kernel(const float* __restrict__ in,
                                  _Float16* __restrict__ out, int n) {
  int i = blockIdx.x * blockDim.x + threadIdx.x;
  if (i < n) out[i] = (_Float16)in[i];
}

// Tiled GEMM body: block tile 128x128, 8 waves (4 M-groups x 2 N-groups),
// wave tile 32x64 -> 8 WMMAs per K-step. Double-buffered async LDS staging.
// c[rr*4 + t] accumulates row-subtile rr (0..1), col-subtile t (0..3).
__device__ __forceinline__ void gemm_body(const _Float16* __restrict__ X,
                                          const _Float16* __restrict__ W,
                                          v8f c[8]) {
  __shared__ __align__(16) _Float16 xs[2][128][32];
  __shared__ __align__(16) _Float16 wt[2][128][32];
  const int tid  = threadIdx.x;
  const int lane = tid & 31;
  const int wv   = tid >> 5;
  const int g    = lane >> 4;
  const int ln   = lane & 15;
  const int i0b  = blockIdx.x * 128;
  const int j0b  = blockIdx.y * 128;
  const int mrow = (wv >> 1) * 32;          // wave's row offset in block tile
  const int ncol = (wv & 1) * 64;           // wave's col offset in block tile

  // copy-chunk geometry (16B chunks): each tile = 512 chunks, 4/row, 2/thread
  const int c0 = tid * 2, c1 = tid * 2 + 1;
  const int r0 = c0 >> 2, o0 = (c0 & 3) * 8;
  const int r1 = c1 >> 2, o1 = (c1 & 3) * 8;

  // prologue: stage k0 = 0 into buffer 0
  async_ld16(X + (size_t)(i0b + r0) * D_MODEL + o0, &xs[0][r0][o0]);
  async_ld16(X + (size_t)(i0b + r1) * D_MODEL + o1, &xs[0][r1][o1]);
  async_ld16(W + (size_t)(j0b + r0) * D_MODEL + o0, &wt[0][r0][o0]);
  async_ld16(W + (size_t)(j0b + r1) * D_MODEL + o1, &wt[0][r1][o1]);

  for (int k0 = 0, it = 0; k0 < D_MODEL; k0 += 32, ++it) {
    const int cur = it & 1;
    wait_async0();
    __syncthreads();                               // buf[cur] visible to all waves
    const int kn = k0 + 32;
    if (kn < D_MODEL) {                            // prefetch next K-slab
      async_ld16(X + (size_t)(i0b + r0) * D_MODEL + kn + o0, &xs[cur ^ 1][r0][o0]);
      async_ld16(X + (size_t)(i0b + r1) * D_MODEL + kn + o1, &xs[cur ^ 1][r1][o1]);
      async_ld16(W + (size_t)(j0b + r0) * D_MODEL + kn + o0, &wt[cur ^ 1][r0][o0]);
      async_ld16(W + (size_t)(j0b + r1) * D_MODEL + kn + o1, &wt[cur ^ 1][r1][o1]);
    }
    // batch all fragment loads, then all WMMAs
    v16h a0 = load_frag_a(&xs[cur][mrow + ln][0], g);
    v16h a1 = load_frag_a(&xs[cur][mrow + 16 + ln][0], g);
    v16h b[4];
    for (int t = 0; t < 4; ++t)
      b[t] = load_frag_b(&wt[cur][ncol + t * 16 + ln][0], g);
    for (int t = 0; t < 4; ++t)
      c[t] = __builtin_amdgcn_wmma_f32_16x16x32_f16(false, a0, false, b[t],
                                                    (short)0, c[t], false, false);
    for (int t = 0; t < 4; ++t)
      c[4 + t] = __builtin_amdgcn_wmma_f32_16x16x32_f16(false, a1, false, b[t],
                                                        (short)0, c[4 + t], false, false);
  }
}

// mode 0: Q -> RoPE, [B,H,S,HD]; 1: K -> RoPE, [B,H,S,HD]; 2: V -> [B,H,HD,S]
__global__ void __launch_bounds__(256)
qkv_gemm_rope(const _Float16* __restrict__ X, const _Float16* __restrict__ W,
              _Float16* __restrict__ dst, int mode) {
  const int lane = threadIdx.x & 31;
  const int wv   = threadIdx.x >> 5;
  const int g    = lane >> 4;
  const int ln   = lane & 15;
  const int i0   = blockIdx.x * 128 + (wv >> 1) * 32;
  const int j0   = blockIdx.y * 128 + (wv & 1) * 64;

  v8f c[8] = {};
  gemm_body(X, W, c);

  const float LN1E4_OVER_32 = 0.28782313662425572f; // ln(10000)/32
  for (int t = 0; t < 4; ++t) {
    const int n = j0 + t * 16 + ln;
    const int h = n >> 6, e = n & 63;
    const float freq = __expf(-(float)(e >> 1) * LN1E4_OVER_32);
    for (int rr = 0; rr < 2; ++rr)
      for (int r = 0; r < 8; ++r) {
        const int m  = i0 + rr * 16 + r + 8 * g;
        const int bi = m >> 11, si = m & (S_LEN - 1);
        float v = c[rr * 4 + t][r];
        if (mode < 2) {
          float sn, cs;
          __sincosf((float)si * freq, &sn, &cs);
          float partner = __shfl_xor(v, 1, 32);   // RoPE pair lives in adjacent lane
          v = (e & 1) ? (v * cs + partner * sn) : (v * cs - partner * sn);
          dst[((size_t)(bi * N_HEADS + h) * S_LEN + si) * HEAD_DIM + e] = (_Float16)v;
        } else {
          dst[((size_t)(bi * N_HEADS + h) * HEAD_DIM + e) * S_LEN + si] = (_Float16)v;
        }
      }
  }
}

// Flash attention: grid (S/64, B*H), 128 thr = 4 waves, one 16-row q-tile each.
// K/V tiles (shared by all 4 waves) are double-buffered in LDS via async copies.
__global__ void __launch_bounds__(128)
flash_attn(const _Float16* __restrict__ Qb, const _Float16* __restrict__ Kb,
           const _Float16* __restrict__ Vt, const unsigned char* __restrict__ mask,
           _Float16* __restrict__ Ob) {
  __shared__ __align__(16) _Float16 kls[2][32][64];  // [key in block][hd]
  __shared__ __align__(16) _Float16 vls[2][64][32];  // [hd][key in block]
  __shared__ __align__(16) _Float16 plds[4][16][32]; // per-wave P staging
  const int tid  = threadIdx.x;
  const int lane = tid & 31;
  const int wv   = tid >> 5;
  const int g    = lane >> 4;
  const int ln   = lane & 15;
  const int bh   = blockIdx.y;
  const int bi   = bh >> 4;
  const int h    = bh & 15;
  const int q0   = (blockIdx.x * 4 + wv) * 16;

  const _Float16* Qp = Qb + ((size_t)bh * S_LEN + q0) * HEAD_DIM;
  const _Float16* Kp = Kb + (size_t)bh * S_LEN * HEAD_DIM;
  const _Float16* Vp = Vt + (size_t)bh * HEAD_DIM * S_LEN;
  const unsigned char* mp = mask + (size_t)bi * S_LEN;

  // copy-chunk geometry: K tile 256 chunks (8/row), V tile 256 chunks (4/row)
  const int ca = tid * 2, cb = tid * 2 + 1;
  const int kr0 = ca >> 3, ko0 = (ca & 7) * 8;
  const int kr1 = cb >> 3, ko1 = (cb & 7) * 8;
  const int vr0 = ca >> 2, vo0 = (ca & 3) * 8;
  const int vr1 = cb >> 2, vo1 = (cb & 3) * 8;

  const v16h aq0 = load_frag_a(Qp + (size_t)ln * HEAD_DIM, g);       // hd 0..31
  const v16h aq1 = load_frag_a(Qp + (size_t)ln * HEAD_DIM + 32, g);  // hd 32..63

  float mrow[8], lrow[8];
  v8f acc[4] = {};
  for (int r = 0; r < 8; ++r) { mrow[r] = -1.0e30f; lrow[r] = 0.0f; }

  // prologue: stage key-block 0 into buffer 0
  async_ld16(Kp + (size_t)kr0 * HEAD_DIM + ko0, &kls[0][kr0][ko0]);
  async_ld16(Kp + (size_t)kr1 * HEAD_DIM + ko1, &kls[0][kr1][ko1]);
  async_ld16(Vp + (size_t)vr0 * S_LEN + vo0,    &vls[0][vr0][vo0]);
  async_ld16(Vp + (size_t)vr1 * S_LEN + vo1,    &vls[0][vr1][vo1]);

  for (int kb = 0, it = 0; kb < S_LEN; kb += 32, ++it) {
    const int cur = it & 1;
    wait_async0();
    __syncthreads();                               // K/V buf[cur] ready for all waves
    const int kn = kb + 32;
    if (kn < S_LEN) {                              // prefetch next key block
      async_ld16(Kp + (size_t)(kn + kr0) * HEAD_DIM + ko0, &kls[cur ^ 1][kr0][ko0]);
      async_ld16(Kp + (size_t)(kn + kr1) * HEAD_DIM + ko1, &kls[cur ^ 1][kr1][ko1]);
      async_ld16(Vp + (size_t)vr0 * S_LEN + kn + vo0,      &vls[cur ^ 1][vr0][vo0]);
      async_ld16(Vp + (size_t)vr1 * S_LEN + kn + vo1,      &vls[cur ^ 1][vr1][vo1]);
    }

    // batch all K and V fragment loads, then the S-WMMAs
    v16h kb0 = load_frag_b(&kls[cur][ln][0], g);
    v16h kb1 = load_frag_b(&kls[cur][ln][32], g);
    v16h kb2 = load_frag_b(&kls[cur][16 + ln][0], g);
    v16h kb3 = load_frag_b(&kls[cur][16 + ln][32], g);
    v16h bv[4];
    for (int t = 0; t < 4; ++t)
      bv[t] = load_frag_b(&vls[cur][t * 16 + ln][0], g);

    v8f s0 = {}, s1 = {};
    s0 = __builtin_amdgcn_wmma_f32_16x16x32_f16(false, aq0, false, kb0, (short)0, s0, false, false);
    s0 = __builtin_amdgcn_wmma_f32_16x16x32_f16(false, aq1, false, kb1, (short)0, s0, false, false);
    s1 = __builtin_amdgcn_wmma_f32_16x16x32_f16(false, aq0, false, kb2, (short)0, s1, false, false);
    s1 = __builtin_amdgcn_wmma_f32_16x16x32_f16(false, aq1, false, kb3, (short)0, s1, false, false);

    const unsigned char mk0 = mp[kb + ln];
    const unsigned char mk1 = mp[kb + 16 + ln];

    for (int r = 0; r < 8; ++r) {
      float v0 = mk0 ? -1.0e9f : s0[r] * 0.125f;   // 1/sqrt(64), then mask
      float v1 = mk1 ? -1.0e9f : s1[r] * 0.125f;
      float mx = fmaxf(v0, v1);
      for (int off = 1; off < 16; off <<= 1) mx = fmaxf(mx, __shfl_xor(mx, off, 32));
      const float mnew  = fmaxf(mrow[r], mx);
      const float alpha = __expf(mrow[r] - mnew);
      const float p0 = __expf(v0 - mnew);
      const float p1 = __expf(v1 - mnew);
      float rs = p0 + p1;
      for (int off = 1; off < 16; off <<= 1) rs += __shfl_xor(rs, off, 32);
      lrow[r] = lrow[r] * alpha + rs;
      mrow[r] = mnew;
      for (int t = 0; t < 4; ++t) acc[t][r] *= alpha;
      // C-layout -> A-layout via per-wave LDS slice (same-wave DS ops in-order;
      // no cross-wave sharing, so no barrier needed)
      plds[wv][r + 8 * g][ln]      = (_Float16)p0;
      plds[wv][r + 8 * g][16 + ln] = (_Float16)p1;
    }
    v16h ap = load_frag_a(&plds[wv][ln][0], g);
    for (int t = 0; t < 4; ++t)
      acc[t] = __builtin_amdgcn_wmma_f32_16x16x32_f16(false, ap, false, bv[t],
                                                      (short)0, acc[t], false, false);
  }

  for (int t = 0; t < 4; ++t)
    for (int r = 0; r < 8; ++r) {
      const int m = q0 + r + 8 * g;
      const size_t row = (size_t)bi * S_LEN + m;
      const int col = h * HEAD_DIM + t * 16 + ln;
      Ob[row * D_MODEL + col] = (_Float16)(acc[t][r] / lrow[r]);
    }
}

// out = O @ Wo^T, fp32 result.
__global__ void __launch_bounds__(256)
out_gemm(const _Float16* __restrict__ O, const _Float16* __restrict__ W,
         float* __restrict__ out) {
  const int lane = threadIdx.x & 31;
  const int wv   = threadIdx.x >> 5;
  const int g    = lane >> 4;
  const int ln   = lane & 15;
  const int i0   = blockIdx.x * 128 + (wv >> 1) * 32;
  const int j0   = blockIdx.y * 128 + (wv & 1) * 64;

  v8f c[8] = {};
  gemm_body(O, W, c);

  for (int t = 0; t < 4; ++t)
    for (int rr = 0; rr < 2; ++rr)
      for (int r = 0; r < 8; ++r) {
        const int m = i0 + rr * 16 + r + 8 * g;
        const int n = j0 + t * 16 + ln;
        out[(size_t)m * D_MODEL + n] = c[rr * 4 + t][r];
      }
}

extern "C" void kernel_launch(void* const* d_in, const int* in_sizes, int n_in,
                              void* d_out, int out_size, void* d_ws, size_t ws_size,
                              hipStream_t stream) {
  (void)in_sizes; (void)n_in; (void)out_size; (void)ws_size;
  const float* x            = (const float*)d_in[0];
  const unsigned char* mask = (const unsigned char*)d_in[1];  // jnp bool_ = 1 byte
  const float* Wq           = (const float*)d_in[2];
  const float* Wk           = (const float*)d_in[3];
  const float* Wv           = (const float*)d_in[4];
  const float* Wo           = (const float*)d_in[5];
  float* out                = (float*)d_out;

  const size_t ND = (size_t)BATCH * S_LEN * D_MODEL;  // 4M elems
  const size_t NW = (size_t)D_MODEL * D_MODEL;        // 1M elems
  _Float16* xb  = (_Float16*)d_ws;
  _Float16* Wqb = xb  + ND;
  _Float16* Wkb = Wqb + NW;
  _Float16* Wvb = Wkb + NW;
  _Float16* Wob = Wvb + NW;
  _Float16* Qb  = Wob + NW;
  _Float16* Kb  = Qb  + ND;
  _Float16* Vt  = Kb  + ND;
  _Float16* Ob  = Vt  + ND;   // total 24M f16 = 48 MB of workspace

  f32_to_f16_kernel<<<(int)((ND + 255) / 256), 256, 0, stream>>>(x,  xb,  (int)ND);
  f32_to_f16_kernel<<<(int)((NW + 255) / 256), 256, 0, stream>>>(Wq, Wqb, (int)NW);
  f32_to_f16_kernel<<<(int)((NW + 255) / 256), 256, 0, stream>>>(Wk, Wkb, (int)NW);
  f32_to_f16_kernel<<<(int)((NW + 255) / 256), 256, 0, stream>>>(Wv, Wvb, (int)NW);
  f32_to_f16_kernel<<<(int)((NW + 255) / 256), 256, 0, stream>>>(Wo, Wob, (int)NW);

  dim3 gp(BATCH * S_LEN / 128, D_MODEL / 128);  // 32 x 8
  qkv_gemm_rope<<<gp, 256, 0, stream>>>(xb, Wqb, Qb, 0);
  qkv_gemm_rope<<<gp, 256, 0, stream>>>(xb, Wkb, Kb, 1);
  qkv_gemm_rope<<<gp, 256, 0, stream>>>(xb, Wvb, Vt, 2);

  dim3 ga(S_LEN / 64, BATCH * N_HEADS);         // 32 x 32
  flash_attn<<<ga, 128, 0, stream>>>(Qb, Kb, Vt, mask, Ob);

  out_gemm<<<gp, 256, 0, stream>>>(Ob, Wob, out);
}
